// STGAT_65369402245373
// MI455X (gfx1250) — matrix operand order repaired and checked by reading, not compiled, not run on hardware
//
#include <hip/hip_runtime.h>
#include <math.h>

#define T_ 12
#define N_ 20000
#define E_ 640000
#define NF_ 24
#define EF_ 12
#define H_ 128
#define HEADS_ 4
#define D_ 32
#define L_ 3
#define NHOR_ 3
#define NQ_ 3
#define CHUNK_ 2000
#define CH_ROWS_ (CHUNK_ * T_)

// LDS row strides (in halves), padded to avoid bank conflicts while keeping
// b128 (16B) alignment for A fragments and 32B alignment for B fragments.
#define AS_STRIDE 136   // 272B/row: bank advance 4/row, 16B-aligned fragments
#define BS_STRIDE 144   // 288B/row: bank advance 8/row, 32B-aligned fragments

typedef _Float16 h8  __attribute__((ext_vector_type(8)));
typedef _Float16 h16 __attribute__((ext_vector_type(16)));
typedef float    f8  __attribute__((ext_vector_type(8)));

// ---------------------------------------------------------------------------
// WMMA GEMM: C[M x (NT*16)] = act(A[M x 128] @ B[128 x NT*16] + bias)
// A fp32 row-major, B fp32 row-major (K x N), C fp32 row-major.
// f16 compute, f32 accumulate via v_wmma_f32_16x16x32_f16.
// ---------------------------------------------------------------------------
template <int NT, int ACT>
__global__ __launch_bounds__(256) void gemm_k128(
    const float* __restrict__ A, const float* __restrict__ B,
    const float* __restrict__ bias, float* __restrict__ C, int M)
{
    constexpr int NC = NT * 16;
    __shared__ __align__(32) _Float16 As[128 * AS_STRIDE];
    __shared__ __align__(32) _Float16 Bs[NC * BS_STRIDE];

    const int tid = threadIdx.x;
    const int m0  = blockIdx.x * 128;

    // prefetch A tile rows (global_prefetch_b8; speculative, OOB-safe)
    {
        int pr = m0 + (tid >> 1);
        __builtin_prefetch(A + (size_t)pr * 128 + (tid & 1) * 64, 0, 0);
    }

    // Stage A tile (128x128) as f16, zero-fill past M.
    for (int e = tid; e < 128 * 32; e += 256) {
        int row = e >> 5;
        int c4  = (e & 31) * 4;
        int gr  = m0 + row;
        float4 v = make_float4(0.f, 0.f, 0.f, 0.f);
        if (gr < M) v = *(const float4*)(A + (size_t)gr * 128 + c4);
        As[row * AS_STRIDE + c4 + 0] = (_Float16)v.x;
        As[row * AS_STRIDE + c4 + 1] = (_Float16)v.y;
        As[row * AS_STRIDE + c4 + 2] = (_Float16)v.z;
        As[row * AS_STRIDE + c4 + 3] = (_Float16)v.w;
    }
    // Stage B transposed: Bs[n][k] = B[k][n] (float4 loads along contiguous N)
    for (int e = tid; e < 128 * (NC / 4); e += 256) {
        int k  = e / (NC / 4);
        int n4 = (e % (NC / 4)) * 4;
        float4 v = *(const float4*)(B + (size_t)k * NC + n4);
        Bs[(n4 + 0) * BS_STRIDE + k] = (_Float16)v.x;
        Bs[(n4 + 1) * BS_STRIDE + k] = (_Float16)v.y;
        Bs[(n4 + 2) * BS_STRIDE + k] = (_Float16)v.z;
        Bs[(n4 + 3) * BS_STRIDE + k] = (_Float16)v.w;
    }
    __syncthreads();

    const int wid   = tid >> 5;
    const int lane  = tid & 31;
    const int lcol  = lane & 15;
    const int lhalf = lane >> 4;
    const int lrow  = wid * 16 + lcol;

    f8 acc[NT] = {};

#pragma unroll
    for (int kb = 0; kb < 4; ++kb) {
        // A fragment (16x32 f16): lanes 0-15 K 0..15, lanes 16-31 K 16..31
        const _Float16* ap = &As[lrow * AS_STRIDE + kb * 32 + lhalf * 16];
        h8 alo = *(const h8*)ap;
        h8 ahi = *(const h8*)(ap + 8);
        h16 afrag = __builtin_shufflevector(alo, ahi, 0, 1, 2, 3, 4, 5, 6, 7,
                                            8, 9, 10, 11, 12, 13, 14, 15);
#pragma unroll
        for (int j = 0; j < NT; ++j) {
            // B fragment (32x16 f16): lanes 0-15 K 0..15, lanes 16-31 K 16..31
            const _Float16* bp =
                &Bs[(j * 16 + lcol) * BS_STRIDE + kb * 32 + lhalf * 16];
            h16 bfrag = *(const h16*)bp;
            acc[j] = __builtin_amdgcn_wmma_f32_16x16x32_f16(
                false, afrag, false, bfrag, (short)0, acc[j], false, false);
        }
    }

#pragma unroll
    for (int j = 0; j < NT; ++j) {
        int col  = j * 16 + lcol;
        float bv = bias ? bias[col] : 0.f;
#pragma unroll
        for (int r = 0; r < 8; ++r) {
            int row = wid * 16 + r + lhalf * 8;
            int gr  = m0 + row;
            if (gr < M) {
                float v = acc[j][r] + bv;
                if (ACT == 1) v = v > 0.f ? v : 0.f;
                C[(size_t)gr * NC + col] = v;
            }
        }
    }
}

// ---------------------------------------------------------------------------
// helpers
// ---------------------------------------------------------------------------
__device__ inline float wave_sum(float v)
{
    for (int o = 16; o > 0; o >>= 1) v += __shfl_xor(v, o, 32);
    return v;
}
__device__ inline unsigned fenc(float f)
{
    unsigned b = __float_as_uint(f);
    return (b & 0x80000000u) ? ~b : (b | 0x80000000u);
}
__device__ inline float fdec(unsigned k)
{
    return (k & 0x80000000u) ? __uint_as_float(k ^ 0x80000000u)
                             : __uint_as_float(~k);
}
__device__ inline float softplusf(float x)
{
    return (x > 20.f) ? x : log1pf(expf(x));
}

__global__ void fill_f32(float* p, float v, int n)
{
    int i = blockIdx.x * 256 + threadIdx.x;
    if (i < n) p[i] = v;
}
__global__ void fill_u32(unsigned* p, unsigned v, int n)
{
    int i = blockIdx.x * 256 + threadIdx.x;
    if (i < n) p[i] = v;
}

// ---------------------------------------------------------------------------
// node embedding: H0 = relu(LN(nf @ node_w + node_b))   one wave per node
// ---------------------------------------------------------------------------
__global__ void embed_kernel(const float* __restrict__ nf,
                             const float* __restrict__ W,
                             const float* __restrict__ b,
                             const float* __restrict__ g,
                             const float* __restrict__ beta,
                             float* __restrict__ H0)
{
    int node = blockIdx.x * 8 + (threadIdx.x >> 5);
    int lane = threadIdx.x & 31;
    if (node >= N_) return;
    const float* x = nf + (size_t)node * NF_;
    float v[4];
#pragma unroll
    for (int k = 0; k < 4; ++k) {
        int i = k * 32 + lane;
        float s = b[i];
        for (int f = 0; f < NF_; ++f) s += x[f] * W[f * H_ + i];
        v[k] = s;
    }
    float mu = wave_sum(v[0] + v[1] + v[2] + v[3]) * (1.f / 128.f);
    float var = 0.f;
#pragma unroll
    for (int k = 0; k < 4; ++k) {
        float d = v[k] - mu;
        var += d * d;
    }
    var = wave_sum(var) * (1.f / 128.f);
    float rs = rsqrtf(var + 1e-5f);
#pragma unroll
    for (int k = 0; k < 4; ++k) {
        int i = k * 32 + lane;
        float o = (v[k] - mu) * rs * g[i] + beta[i];
        H0[(size_t)node * H_ + i] = o > 0.f ? o : 0.f;
    }
}

// W2[l][f][h] = sum_d We_l[f, h*32+d] * a_edge_l[h, d]
__global__ void w2_kernel(const float* We0, const float* We1, const float* We2,
                          const float* ae0, const float* ae1, const float* ae2,
                          float* __restrict__ W2)
{
    int idx = blockIdx.x * 256 + threadIdx.x;
    if (idx >= L_ * EF_ * HEADS_) return;
    int l = idx / (EF_ * HEADS_);
    int f = (idx / HEADS_) % EF_;
    int h = idx % HEADS_;
    const float* We = (l == 0) ? We0 : (l == 1) ? We1 : We2;
    const float* ae = (l == 0) ? ae0 : (l == 1) ? ae1 : ae2;
    float s = 0.f;
    for (int d = 0; d < D_; ++d) s += We[f * H_ + h * D_ + d] * ae[h * D_ + d];
    W2[idx] = s;
}

// ATERM[l][e][h] = sum_f ef[e][f] * W2[l][f][h]
__global__ void aterm_kernel(const float* __restrict__ ef,
                             const float* __restrict__ W2,
                             float* __restrict__ ATERM)
{
    int e = blockIdx.x * 256 + threadIdx.x;
    if (e >= E_) return;
    float x[EF_];
#pragma unroll
    for (int f = 0; f < EF_; ++f) x[f] = ef[(size_t)e * EF_ + f];
#pragma unroll
    for (int l = 0; l < L_; ++l)
#pragma unroll
        for (int h = 0; h < HEADS_; ++h) {
            float s = 0.f;
#pragma unroll
            for (int f = 0; f < EF_; ++f)
                s += x[f] * W2[l * EF_ * HEADS_ + f * HEADS_ + h];
            ATERM[(size_t)l * E_ * HEADS_ + (size_t)e * HEADS_ + h] = s;
        }
}

// a_src / a_dst projections: AS[n,h] = sum_d HH[n,h*32+d]*a_src[h,d]
__global__ void asad_kernel(const float* __restrict__ HH,
                            const float* __restrict__ a_src,
                            const float* __restrict__ a_dst,
                            float* __restrict__ AS, float* __restrict__ AD)
{
    int n = blockIdx.x * 256 + threadIdx.x;
    if (n >= N_) return;
    const float* hp = HH + (size_t)n * H_;
#pragma unroll
    for (int h = 0; h < HEADS_; ++h) {
        float sa = 0.f, sd = 0.f;
        for (int d = 0; d < D_; ++d) {
            float hv = hp[h * D_ + d];
            sa += hv * a_src[h * D_ + d];
            sd += hv * a_dst[h * D_ + d];
        }
        AS[n * HEADS_ + h] = sa;
        AD[n * HEADS_ + h] = sd;
    }
}

// pass 1: alpha = leaky_relu(a_s[src]+a_d[dst]+aterm); segment max via atomics
__global__ void gat_pass1(const int* __restrict__ src, const int* __restrict__ dst,
                          const float* __restrict__ AS, const float* __restrict__ AD,
                          const float* __restrict__ aterm,
                          float* __restrict__ ALPHA, unsigned* __restrict__ MB)
{
    int e = blockIdx.x * 256 + threadIdx.x;
    if (e >= E_) return;
    int s = src[e], d = dst[e];
#pragma unroll
    for (int h = 0; h < HEADS_; ++h) {
        float a = AS[s * HEADS_ + h] + AD[d * HEADS_ + h] +
                  aterm[(size_t)e * HEADS_ + h];
        a = a > 0.f ? a : 0.2f * a;
        ALPHA[(size_t)e * HEADS_ + h] = a;
        atomicMax(&MB[d * HEADS_ + h], fenc(a));
    }
}

// pass 2: w = exp(alpha - m); scatter-add denom and weighted features
__global__ void gat_pass2(const int* __restrict__ src, const int* __restrict__ dst,
                          const float* __restrict__ ALPHA,
                          const unsigned* __restrict__ MB,
                          const float* __restrict__ HH,
                          float* __restrict__ DEN, float* __restrict__ AGG)
{
    int e = blockIdx.x * 256 + threadIdx.x;
    if (e >= E_) return;
    int s = src[e], d = dst[e];
#pragma unroll
    for (int h = 0; h < HEADS_; ++h) {
        float m = fdec(MB[d * HEADS_ + h]);
        float w = expf(ALPHA[(size_t)e * HEADS_ + h] - m);
        atomicAdd(&DEN[d * HEADS_ + h], w);
        const float4* hp = (const float4*)(HH + (size_t)s * H_ + h * D_);
        float* op = AGG + (size_t)d * H_ + h * D_;
#pragma unroll
        for (int q = 0; q < 8; ++q) {
            float4 hv = hp[q];
            atomicAdd(op + q * 4 + 0, hv.x * w);
            atomicAdd(op + q * 4 + 1, hv.y * w);
            atomicAdd(op + q * 4 + 2, hv.z * w);
            atomicAdd(op + q * 4 + 3, hv.w * w);
        }
    }
}

// H0 = LN(H0 + relu(AGG/(DEN+1e-16) + b))   one wave per node
__global__ void gat_update(float* __restrict__ H0, const float* __restrict__ AGG,
                           const float* __restrict__ DEN,
                           const float* __restrict__ bias,
                           const float* __restrict__ g, const float* __restrict__ b)
{
    int node = blockIdx.x * 8 + (threadIdx.x >> 5);
    int lane = threadIdx.x & 31;
    if (node >= N_) return;
    float v[4];
#pragma unroll
    for (int k = 0; k < 4; ++k) {
        int i = k * 32 + lane;      // head = k
        float hn = AGG[(size_t)node * H_ + i] / (DEN[node * HEADS_ + k] + 1e-16f) +
                   bias[i];
        hn = hn > 0.f ? hn : 0.f;
        v[k] = H0[(size_t)node * H_ + i] + hn;
    }
    float mu = wave_sum(v[0] + v[1] + v[2] + v[3]) * (1.f / 128.f);
    float var = 0.f;
#pragma unroll
    for (int k = 0; k < 4; ++k) {
        float d = v[k] - mu;
        var += d * d;
    }
    var = wave_sum(var) * (1.f / 128.f);
    float rs = rsqrtf(var + 1e-5f);
#pragma unroll
    for (int k = 0; k < 4; ++k) {
        int i = k * 32 + lane;
        H0[(size_t)node * H_ + i] = (v[k] - mu) * rs * g[i] + b[i];
    }
}

// XT[n, t, :] = H0[n, :] + pe[t, :]
__global__ void xt_write(const float* __restrict__ H0, float* __restrict__ XT, int t)
{
    int idx = blockIdx.x * 256 + threadIdx.x;
    if (idx >= N_ * H_) return;
    int n = idx / H_, i = idx % H_;
    int j = i >> 1;
    float ang = (float)t * __expf(-(float)(2 * j) * (logf(10000.f) / (float)H_));
    float pe = (i & 1) ? cosf(ang) : sinf(ang);
    XT[((size_t)n * T_ + t) * H_ + i] = H0[idx] + pe;
}

// per-node temporal attention over T=12, one thread per (node, head, qt)
__global__ void attn_kernel(const float* __restrict__ Q, const float* __restrict__ K,
                            const float* __restrict__ V, float* __restrict__ O)
{
    int idx = blockIdx.x * 256 + threadIdx.x;
    if (idx >= CHUNK_ * HEADS_ * T_) return;
    int qt = idx % T_;
    int h  = (idx / T_) % HEADS_;
    int nl = idx / (T_ * HEADS_);

    const float* q = Q + ((size_t)(nl * T_ + qt) * H_ + h * D_);
    float qv[D_];
#pragma unroll
    for (int d = 0; d < D_; ++d) qv[d] = q[d];

    float sc[T_];
    float mx = -1e30f;
#pragma unroll
    for (int kt = 0; kt < T_; ++kt) {
        const float* kk = K + ((size_t)(nl * T_ + kt) * H_ + h * D_);
        float s = 0.f;
#pragma unroll
        for (int d = 0; d < D_; ++d) s += qv[d] * kk[d];
        s *= 0.17677669529663687f; // 1/sqrt(32)
        sc[kt] = s;
        mx = fmaxf(mx, s);
    }
    float den = 0.f;
#pragma unroll
    for (int kt = 0; kt < T_; ++kt) {
        sc[kt] = expf(sc[kt] - mx);
        den += sc[kt];
    }
    float inv = 1.f / den;
    float o[D_] = {};
#pragma unroll
    for (int kt = 0; kt < T_; ++kt) {
        float w = sc[kt] * inv;
        const float* vv = V + ((size_t)(nl * T_ + kt) * H_ + h * D_);
#pragma unroll
        for (int d = 0; d < D_; ++d) o[d] += w * vv[d];
    }
    float* op = O + ((size_t)(nl * T_ + qt) * H_ + h * D_);
#pragma unroll
    for (int d = 0; d < D_; ++d) op[d] = o[d];
}

// x = LN(x + add), one wave per row of 128
__global__ void ln_add_kernel(float* __restrict__ x, const float* __restrict__ add,
                              const float* __restrict__ g, const float* __restrict__ b,
                              int rows)
{
    int r = blockIdx.x * 8 + (threadIdx.x >> 5);
    int lane = threadIdx.x & 31;
    if (r >= rows) return;
    float v[4];
#pragma unroll
    for (int k = 0; k < 4; ++k) {
        int i = k * 32 + lane;
        v[k] = x[(size_t)r * H_ + i] + add[(size_t)r * H_ + i];
    }
    float mu = wave_sum(v[0] + v[1] + v[2] + v[3]) * (1.f / 128.f);
    float var = 0.f;
#pragma unroll
    for (int k = 0; k < 4; ++k) {
        float d = v[k] - mu;
        var += d * d;
    }
    var = wave_sum(var) * (1.f / 128.f);
    float rs = rsqrtf(var + 1e-5f);
#pragma unroll
    for (int k = 0; k < 4; ++k) {
        int i = k * 32 + lane;
        x[(size_t)r * H_ + i] = (v[k] - mu) * rs * g[i] + b[i];
    }
}

__global__ void xlast_kernel(const float* __restrict__ XT, float* __restrict__ XL)
{
    int idx = blockIdx.x * 256 + threadIdx.x;
    if (idx >= N_ * H_) return;
    int n = idx / H_, i = idx % H_;
    XL[idx] = XT[((size_t)n * T_ + (T_ - 1)) * H_ + i];
}

__global__ void logit_kernel(const float* __restrict__ T1,
                             const float* __restrict__ dW2,
                             const float* __restrict__ db2, float* __restrict__ out)
{
    int n = blockIdx.x * 256 + threadIdx.x;
    if (n >= N_) return;
    float s = db2[0];
    for (int j = 0; j < 64; ++j) s += T1[(size_t)n * 64 + j] * dW2[j];
    out[n] = 1.f / (1.f + expf(-s));
}

__global__ void qnt_kernel(const float* __restrict__ HB,
                           const float* __restrict__ qW,
                           const float* __restrict__ qb, float* __restrict__ outbase)
{
    int n = blockIdx.x * 256 + threadIdx.x;
    if (n >= N_) return;
    float v[2 * NQ_];
#pragma unroll
    for (int c = 0; c < 2 * NQ_; ++c) v[c] = qb[c];
    const float* hp = HB + (size_t)n * H_;
    for (int i = 0; i < H_; ++i) {
        float hv = hp[i];
#pragma unroll
        for (int c = 0; c < 2 * NQ_; ++c) v[c] += hv * qW[i * 2 * NQ_ + c];
    }
#pragma unroll
    for (int j = 0; j < NQ_; ++j) {
        outbase[(size_t)N_ + (size_t)n * NQ_ + j]     = softplusf(v[j]);
        outbase[(size_t)4 * N_ + (size_t)n * NQ_ + j] = 1.f + softplusf(v[NQ_ + j]);
    }
}

// ---------------------------------------------------------------------------
// host side
// ---------------------------------------------------------------------------
struct GatP { const float *Wx, *We, *a_src, *a_dst, *a_edge, *b, *ln_g, *ln_b; };
struct TmpP { const float *Wq, *Wk, *Wv, *Wo, *bq, *bk, *bv, *bo, *ln_g, *ln_b; };
struct HorP { const float *W, *b, *dW1, *db1, *dW2, *db2, *qW, *qb; };

static inline int cdiv(long a, long b) { return (int)((a + b - 1) / b); }

extern "C" void kernel_launch(void* const* d_in, const int* in_sizes, int n_in,
                              void* d_out, int out_size, void* d_ws, size_t ws_size,
                              hipStream_t stream)
{
    const float* node_features;
    const int*   edge_index;
    const float* edge_features;
    const float *node_w, *node_b, *node_g, *node_beta;
    GatP gat[L_];
    TmpP tmp[2];
    HorP hor[NHOR_];

    auto F = [&](int i) { return (const float*)d_in[i]; };

    if (in_sizes[0] == T_ * N_ * NF_) {
        // insertion-order flattening
        node_features = F(0);
        edge_index    = (const int*)d_in[1];
        edge_features = F(2);
        node_w = F(3); node_b = F(4); node_g = F(5); node_beta = F(6);
        for (int l = 0; l < L_; ++l) {
            int s = 7 + 8 * l;
            gat[l] = { F(s), F(s + 1), F(s + 2), F(s + 3), F(s + 4), F(s + 5),
                       F(s + 6), F(s + 7) };
        }
        for (int p = 0; p < 2; ++p) {
            int s = 31 + 10 * p;
            tmp[p] = { F(s), F(s + 1), F(s + 2), F(s + 3), F(s + 4), F(s + 5),
                       F(s + 6), F(s + 7), F(s + 8), F(s + 9) };
        }
        for (int p = 0; p < NHOR_; ++p) {
            int s = 51 + 8 * p;
            hor[p] = { F(s), F(s + 1), F(s + 2), F(s + 3), F(s + 4), F(s + 5),
                       F(s + 6), F(s + 7) };
        }
    } else {
        // sorted-key pytree flattening
        edge_features = F(0);
        edge_index    = (const int*)d_in[1];
        node_features = F(2);
        for (int l = 0; l < L_; ++l) {
            int s = 3 + 8 * l; // We, Wx, a_dst, a_edge, a_src, b, ln_b, ln_g
            gat[l].We = F(s);     gat[l].Wx = F(s + 1);
            gat[l].a_dst = F(s + 2); gat[l].a_edge = F(s + 3); gat[l].a_src = F(s + 4);
            gat[l].b = F(s + 5);  gat[l].ln_b = F(s + 6); gat[l].ln_g = F(s + 7);
        }
        for (int p = 0; p < NHOR_; ++p) {
            int s = 27 + 8 * p; // W, b, dW1, dW2, db1, db2, qW, qb
            hor[p].W = F(s); hor[p].b = F(s + 1);
            hor[p].dW1 = F(s + 2); hor[p].dW2 = F(s + 3);
            hor[p].db1 = F(s + 4); hor[p].db2 = F(s + 5);
            hor[p].qW = F(s + 6);  hor[p].qb = F(s + 7);
        }
        node_b = F(51); node_beta = F(52); node_g = F(53); node_w = F(54);
        for (int p = 0; p < 2; ++p) {
            int s = 55 + 10 * p; // Wk, Wo, Wq, Wv, bk, bo, bq, bv, ln_b, ln_g
            tmp[p].Wk = F(s);     tmp[p].Wo = F(s + 1);
            tmp[p].Wq = F(s + 2); tmp[p].Wv = F(s + 3);
            tmp[p].bk = F(s + 4); tmp[p].bo = F(s + 5);
            tmp[p].bq = F(s + 6); tmp[p].bv = F(s + 7);
            tmp[p].ln_b = F(s + 8); tmp[p].ln_g = F(s + 9);
        }
    }

    // workspace carve-up (floats)
    float* w = (float*)d_ws;
    size_t off = 0;
    auto alloc = [&](size_t n) { float* p = w + off; off += n; return p; };

    float* XT    = alloc((size_t)N_ * T_ * H_);
    float* H0    = alloc((size_t)N_ * H_);
    float* HH    = alloc((size_t)N_ * H_);
    float* AGG   = alloc((size_t)N_ * H_);
    float* ATERM = alloc((size_t)L_ * E_ * HEADS_);
    float* ALPHA = alloc((size_t)E_ * HEADS_);
    float* ASb   = alloc((size_t)N_ * HEADS_);
    float* ADb   = alloc((size_t)N_ * HEADS_);
    unsigned* MB = (unsigned*)alloc((size_t)N_ * HEADS_);
    float* DEN   = alloc((size_t)N_ * HEADS_);
    float* W2    = alloc(L_ * EF_ * HEADS_);
    float* QB    = alloc((size_t)CH_ROWS_ * H_);
    float* KB    = alloc((size_t)CH_ROWS_ * H_);
    float* VB    = alloc((size_t)CH_ROWS_ * H_);
    float* OB    = alloc((size_t)CH_ROWS_ * H_);
    float* O2    = alloc((size_t)CH_ROWS_ * H_);
    float* XL    = alloc((size_t)N_ * H_);
    float* HB    = alloc((size_t)N_ * H_);
    float* T1    = alloc((size_t)N_ * 64);

    float* out = (float*)d_out;

    const int* src = edge_index;
    const int* dst = edge_index + E_;

    // edge attention terms (all 3 layers)
    w2_kernel<<<1, 256, 0, stream>>>(gat[0].We, gat[1].We, gat[2].We,
                                     gat[0].a_edge, gat[1].a_edge, gat[2].a_edge, W2);
    aterm_kernel<<<cdiv(E_, 256), 256, 0, stream>>>(edge_features, W2, ATERM);

    // -------- spatial GAT, per timestep (timesteps independent) --------
    for (int t = 0; t < T_; ++t) {
        embed_kernel<<<cdiv(N_, 8), 256, 0, stream>>>(
            node_features + (size_t)t * N_ * NF_, node_w, node_b, node_g,
            node_beta, H0);
        for (int l = 0; l < L_; ++l) {
            gemm_k128<8, 0><<<cdiv(N_, 128), 256, 0, stream>>>(
                H0, gat[l].Wx, (const float*)nullptr, HH, N_);
            asad_kernel<<<cdiv(N_, 256), 256, 0, stream>>>(
                HH, gat[l].a_src, gat[l].a_dst, ASb, ADb);
            fill_u32<<<cdiv(N_ * HEADS_, 256), 256, 0, stream>>>(MB, 0u, N_ * HEADS_);
            fill_f32<<<cdiv(N_ * HEADS_, 256), 256, 0, stream>>>(DEN, 0.f, N_ * HEADS_);
            fill_f32<<<cdiv(N_ * H_, 256), 256, 0, stream>>>(AGG, 0.f, N_ * H_);
            gat_pass1<<<cdiv(E_, 256), 256, 0, stream>>>(
                src, dst, ASb, ADb, ATERM + (size_t)l * E_ * HEADS_, ALPHA, MB);
            gat_pass2<<<cdiv(E_, 256), 256, 0, stream>>>(
                src, dst, ALPHA, MB, HH, DEN, AGG);
            gat_update<<<cdiv(N_, 8), 256, 0, stream>>>(
                H0, AGG, DEN, gat[l].b, gat[l].ln_g, gat[l].ln_b);
        }
        xt_write<<<cdiv(N_ * H_, 256), 256, 0, stream>>>(H0, XT, t);
    }

    // -------- temporal self-attention (per-node, chunked) --------
    for (int p = 0; p < 2; ++p) {
        for (int c = 0; c < N_ / CHUNK_; ++c) {
            float* xtc = XT + (size_t)c * CHUNK_ * T_ * H_;
            gemm_k128<8, 0><<<cdiv(CH_ROWS_, 128), 256, 0, stream>>>(
                xtc, tmp[p].Wq, tmp[p].bq, QB, CH_ROWS_);
            gemm_k128<8, 0><<<cdiv(CH_ROWS_, 128), 256, 0, stream>>>(
                xtc, tmp[p].Wk, tmp[p].bk, KB, CH_ROWS_);
            gemm_k128<8, 0><<<cdiv(CH_ROWS_, 128), 256, 0, stream>>>(
                xtc, tmp[p].Wv, tmp[p].bv, VB, CH_ROWS_);
            attn_kernel<<<cdiv(CHUNK_ * HEADS_ * T_, 256), 256, 0, stream>>>(
                QB, KB, VB, OB);
            gemm_k128<8, 0><<<cdiv(CH_ROWS_, 128), 256, 0, stream>>>(
                OB, tmp[p].Wo, tmp[p].bo, O2, CH_ROWS_);
            ln_add_kernel<<<cdiv(CH_ROWS_, 8), 256, 0, stream>>>(
                xtc, O2, tmp[p].ln_g, tmp[p].ln_b, CH_ROWS_);
        }
    }

    // -------- output heads --------
    xlast_kernel<<<cdiv(N_ * H_, 256), 256, 0, stream>>>(XT, XL);
    for (int p = 0; p < NHOR_; ++p) {
        float* base = out + (size_t)p * 7 * N_;
        gemm_k128<8, 1><<<cdiv(N_, 128), 256, 0, stream>>>(
            XL, hor[p].W, hor[p].b, HB, N_);
        gemm_k128<4, 1><<<cdiv(N_, 128), 256, 0, stream>>>(
            HB, hor[p].dW1, hor[p].db1, T1, N_);
        logit_kernel<<<cdiv(N_, 256), 256, 0, stream>>>(T1, hor[p].dW2, hor[p].db2,
                                                        base);
        qnt_kernel<<<cdiv(N_, 256), 256, 0, stream>>>(HB, hor[p].qW, hor[p].qb, base);
    }

    (void)n_in; (void)out_size; (void)ws_size;
}
